// MultiScaleFeaStNet_14242111553639
// MI455X (gfx1250) — matrix operand order repaired and checked by reading, not compiled, not run on hardware
//
#include <hip/hip_runtime.h>
#include <hip/hip_bf16.h>

// ---------------------------------------------------------------------------
// MultiScaleFeaStNet for MI455X (gfx1250, wave32, WMMA)
//
// Dominant cost: [40000,256] @ [256,6890] = 141 GFLOP  ->  bf16 WMMA,
// fused with relu + [6890,1] head + sigmoid so the 1.1GB intermediate
// never touches memory. l2W (3.5MB bf16) prepacked into WMMA B-fragment
// layout, staged tile-by-tile through LDS (double buffered) so the 8 waves
// of a workgroup share each fragment fetch: global/L2 traffic drops 8x.
// ---------------------------------------------------------------------------

#define HEADS 4
#define N1 40000
#define N2C 20000
#define N3C 10000
#define E1 640000
#define E2C 320000
#define E3C 160000
#define NCOLT 431          // ceil(6890/16) column tiles of layer-2 GEMM
#define NCOLP (NCOLT * 16) // 6896 padded columns
#define NROWT 2500         // 40000 / 16 row tiles
#define WAVES_PER_BLOCK 8
#define TILE_USHORTS 4096  // 8 ksteps * 32 lanes * 16 bf16 = 8KB per column tile

#define CDIV(a, b) (((a) + (b) - 1) / (b))

typedef __attribute__((ext_vector_type(16))) __bf16 v16bf;
typedef __attribute__((ext_vector_type(8)))  float  v8f;

union BFrag {
    v16bf v;
    int4  q[2];
};

__device__ __forceinline__ unsigned short f2bf(float f) {
    unsigned int u = __float_as_uint(f);
    u += 0x7FFFu + ((u >> 16) & 1u);   // round-to-nearest-even
    return (unsigned short)(u >> 16);
}

// ------------------------- FeaStConv building blocks -----------------------

// m[i, o] = sum_k x[i,k] * W[k,o]   (o over HEADS*fo)
__global__ void node_m_kernel(const float* __restrict__ x, const float* __restrict__ W,
                              float* __restrict__ m, int n, int fi, int fop) {
    int idx = blockIdx.x * blockDim.x + threadIdx.x;
    if (idx >= n * fop) return;
    int i = idx / fop, o = idx - i * fop;
    float s = 0.f;
    for (int k = 0; k < fi; ++k) s += x[i * fi + k] * W[k * fop + o];
    m[idx] = s;
}

// u[i, h] = sum_k x[i,k] * U[k,h]
__global__ void node_u_kernel(const float* __restrict__ x, const float* __restrict__ U,
                              float* __restrict__ u, int n, int fi) {
    int idx = blockIdx.x * blockDim.x + threadIdx.x;
    if (idx >= n * HEADS) return;
    int i = idx >> 2, h = idx & 3;
    float s = 0.f;
    for (int k = 0; k < fi; ++k) s += x[i * fi + k] * U[k * HEADS + h];
    u[idx] = s;
}

// degree (incl. self loop) per destination
__global__ void count_kernel(const int* __restrict__ ei, float* __restrict__ cnt,
                             int e, int n) {
    int idx = blockIdx.x * blockDim.x + threadIdx.x;
    if (idx >= e + n) return;
    int d = (idx < e) ? ei[e + idx] : (idx - e);
    atomicAdd(&cnt[d], 1.f);
}

// per-(edge, out-feature): head softmax attention + scatter-add message
__global__ void edge_msg_kernel(const int* __restrict__ ei,
                                const float* __restrict__ m, const float* __restrict__ u,
                                const float* __restrict__ c, float* __restrict__ agg,
                                int e, int n, int fo) {
    int idx = blockIdx.x * blockDim.x + threadIdx.x;
    if (idx >= (e + n) * fo) return;
    int edge = idx / fo, o = idx - edge * fo;
    int s, d;
    if (edge < e) { s = ei[edge]; d = ei[e + edge]; }
    else          { s = edge - e; d = s; }
    const float4 us = ((const float4*)u)[s];
    const float4 ud = ((const float4*)u)[d];
    float l0 = us.x - ud.x + c[0];
    float l1 = us.y - ud.y + c[1];
    float l2 = us.z - ud.z + c[2];
    float l3 = us.w - ud.w + c[3];
    float mx = fmaxf(fmaxf(l0, l1), fmaxf(l2, l3));
    float e0 = __expf(l0 - mx), e1 = __expf(l1 - mx);
    float e2 = __expf(l2 - mx), e3 = __expf(l3 - mx);
    float inv = 1.f / (e0 + e1 + e2 + e3);
    const float* ms = m + (size_t)s * (HEADS * fo);
    float msg = (e0 * ms[o] + e1 * ms[fo + o] + e2 * ms[2 * fo + o] + e3 * ms[3 * fo + o]) * inv;
    atomicAdd(&agg[d * fo + o], msg);
}

// out = relu(agg / cnt + b)
__global__ void finalize_kernel(const float* __restrict__ agg, const float* __restrict__ cnt,
                                const float* __restrict__ b, float* __restrict__ out,
                                int n, int fo) {
    int idx = blockIdx.x * blockDim.x + threadIdx.x;
    if (idx >= n * fo) return;
    int i = idx / fo, o = idx - i * fo;
    float v = agg[idx] / cnt[i] + b[o];
    out[idx] = v > 0.f ? v : 0.f;
}

// graclus max pool: all inputs are post-relu (>=0) so int-bit atomicMax is exact
__global__ void seg_max_kernel(const float* __restrict__ x, const int* __restrict__ cl,
                               float* __restrict__ out, int n, int f) {
    int idx = blockIdx.x * blockDim.x + threadIdx.x;
    if (idx >= n * f) return;
    int i = idx / f, j = idx - i * f;
    atomicMax((int*)&out[cl[i] * f + j], __float_as_int(x[idx]));
}

// out[i] = [a[i, :fa] | bsrc[cl[i], :fb]]
__global__ void concat_up_kernel(const float* __restrict__ a, const float* __restrict__ bsrc,
                                 const int* __restrict__ cl, float* __restrict__ out,
                                 int n, int fa, int fb) {
    int fc = fa + fb;
    int idx = blockIdx.x * blockDim.x + threadIdx.x;
    if (idx >= n * fc) return;
    int i = idx / fc, j = idx - i * fc;
    out[idx] = (j < fa) ? a[i * fa + j] : bsrc[cl[i] * fb + (j - fa)];
}

// --------------------------------- MLP -------------------------------------

// a1 = relu(h @ l1W + l1b) stored bf16 row-major [N1, 256]
__global__ void mlp1_kernel(const float* __restrict__ h, const float* __restrict__ W,
                            const float* __restrict__ b, unsigned short* __restrict__ a1) {
    int idx = blockIdx.x * blockDim.x + threadIdx.x;
    if (idx >= N1 * 256) return;
    int r = idx >> 8, c = idx & 255;
    float s = b[c];
    const float* hr = h + r * 32;
#pragma unroll
    for (int k = 0; k < 32; ++k) s += hr[k] * W[k * 256 + c];
    a1[idx] = f2bf(s > 0.f ? s : 0.f);
}

// prepack l2W (f32, row-major [256,6890]) into WMMA bf16 B-fragment layout:
// Bp[((t*8 + s)*32 + lane)*16 + j] : lane column n = t*16+(lane&15),
//   K = s*32 + ((lane>=16)?8:0) + (j&7) + ((j>=8)?16:0)  (j = 0..15, table order per lane)
__global__ void pack_b_kernel(const float* __restrict__ W, unsigned short* __restrict__ Bp) {
    int idx = blockIdx.x * blockDim.x + threadIdx.x;
    if (idx >= NCOLT * 8 * 32 * 16) return;
    int j    = idx & 15;
    int lane = (idx >> 4) & 31;
    int fs   = idx >> 9;
    int s    = fs & 7;
    int t    = fs >> 3;
    int k    = s * 32 + ((lane >> 4) << 3) + (j & 7) + ((j >> 3) << 4);
    int n    = t * 16 + (lane & 15);
    float v  = (n < 6890) ? W[k * 6890 + n] : 0.f;
    Bp[idx]  = f2bf(v);
}

// pad l2b / oW to 6896 (oW tail zero => padded columns contribute nothing)
__global__ void pack_vec_kernel(const float* __restrict__ b, const float* __restrict__ w,
                                float* __restrict__ bp, float* __restrict__ wp) {
    int i = blockIdx.x * blockDim.x + threadIdx.x;
    if (i >= NCOLP) return;
    bp[i] = (i < 6890) ? b[i] : 0.f;
    wp[i] = (i < 6890) ? w[i] : 0.f;
}

// Fused: relu(a1 @ l2W + l2b) . oW + ob -> sigmoid.
// 8 waves/block, one 16-row tile per wave; B column-tiles staged through LDS
// (double buffered) so all 8 waves share each fragment fetch. Out-of-range
// waves clamp their tile and skip the store so barriers stay uniform.
__global__ __launch_bounds__(256)
void fused_mlp2_kernel(const unsigned short* __restrict__ a1,
                       const unsigned short* __restrict__ Bp,
                       const float* __restrict__ biasp, const float* __restrict__ owp,
                       const float* __restrict__ obp, float* __restrict__ out) {
    __shared__ unsigned short ldsB[2][TILE_USHORTS];   // 2 x 8KB

    int waveInBlk = threadIdx.x >> 5;
    int lane      = threadIdx.x & 31;
    int gwave     = blockIdx.x * WAVES_PER_BLOCK + waveInBlk;
    bool valid    = gwave < NROWT;
    int  tileRow  = valid ? gwave : (NROWT - 1);       // clamp: keep EXEC full

    // A-matrix 16x32 bf16 layout (ISA 7.12.2): lane<16 -> K 0..7 & 16..23,
    // lane>=16 -> K 8..15 & 24..31, row M = lane&15.
    int row = tileRow * 16 + (lane & 15);
    int kb  = (lane >> 4) << 3;                        // 0 or 8
    const unsigned short* ar = a1 + (size_t)row * 256;
    BFrag A[8];
#pragma unroll
    for (int s = 0; s < 8; ++s) {
        int k0 = s * 32 + kb;
        A[s].q[0] = *(const int4*)(ar + k0);
        A[s].q[1] = *(const int4*)(ar + k0 + 16);
    }

    float racc[8];
#pragma unroll
    for (int r = 0; r < 8; ++r) racc[r] = 0.f;

    // cooperative tile loader: 256 threads x 2 int4 = 8KB
    auto load_tile = [&](int t, int buf) {
        const int4* src = (const int4*)(Bp + (size_t)t * TILE_USHORTS);
        int4* dst = (int4*)ldsB[buf];
        dst[threadIdx.x]       = src[threadIdx.x];
        dst[threadIdx.x + 256] = src[threadIdx.x + 256];
    };

    load_tile(0, 0);

    for (int t = 0; t < NCOLT; ++t) {
        __syncthreads();                                // tile t resident; t-1 consumed
        if (t + 1 < NCOLT) load_tile(t + 1, (t + 1) & 1);

        const unsigned short* bt = &ldsB[t & 1][lane * 16];
        v8f acc = {};
#pragma unroll
        for (int s = 0; s < 8; ++s) {
            BFrag B;
            const int4* p = (const int4*)(bt + s * (32 * 16));
            B.q[0] = p[0];
            B.q[1] = p[1];
            acc = __builtin_amdgcn_wmma_f32_16x16x32_bf16(
                false, A[s].v, false, B.v, (short)0, acc, false, false);
        }
        // D layout: lanes 0-15 VGPR r -> (M=r, N=lane); lanes 16-31 -> (M=8+r, N=lane-16)
        int n   = t * 16 + (lane & 15);
        float b = biasp[n];
        float w = owp[n];
#pragma unroll
        for (int r = 0; r < 8; ++r) {
            float v = acc[r] + b;
            racc[r] += (v > 0.f ? v : 0.f) * w;
        }
    }

    // reduce the 16 columns held by each half-wave
#pragma unroll
    for (int r = 0; r < 8; ++r) {
        float v = racc[r];
        v += __shfl_xor(v, 1, 16);
        v += __shfl_xor(v, 2, 16);
        v += __shfl_xor(v, 4, 16);
        v += __shfl_xor(v, 8, 16);
        racc[r] = v;
    }
    if (valid && (lane & 15) == 0) {
        float ob = obp[0];
        int mbase = tileRow * 16 + ((lane >> 4) << 3); // lane 0 -> rows 0..7, lane 16 -> 8..15
#pragma unroll
        for (int r = 0; r < 8; ++r) {
            float s = racc[r] + ob;
            out[mbase + r] = 1.f / (1.f + __expf(-s));
        }
    }
}

// ------------------------------ host driver --------------------------------

static void run_conv(const float* x, int n, int fi, int fo,
                     const int* ei, int e,
                     const float* W, const float* U, const float* c, const float* b,
                     float* out, float* m, float* u, float* agg, float* cnt,
                     hipStream_t stream) {
    int fop = HEADS * fo;
    hipMemsetAsync(agg, 0, (size_t)n * fo * sizeof(float), stream);
    hipMemsetAsync(cnt, 0, (size_t)n * sizeof(float), stream);
    node_m_kernel<<<CDIV(n * fop, 256), 256, 0, stream>>>(x, W, m, n, fi, fop);
    node_u_kernel<<<CDIV(n * HEADS, 256), 256, 0, stream>>>(x, U, u, n, fi);
    count_kernel<<<CDIV(e + n, 256), 256, 0, stream>>>(ei, cnt, e, n);
    edge_msg_kernel<<<CDIV((e + n) * fo, 256), 256, 0, stream>>>(ei, m, u, c, agg, e, n, fo);
    finalize_kernel<<<CDIV(n * fo, 256), 256, 0, stream>>>(agg, cnt, b, out, n, fo);
}

extern "C" void kernel_launch(void* const* d_in, const int* in_sizes, int n_in,
                              void* d_out, int out_size, void* d_ws, size_t ws_size,
                              hipStream_t stream) {
    const float* x        = (const float*)d_in[0];
    const int*   ei1      = (const int*)d_in[1];
    const int*   ei2      = (const int*)d_in[2];
    const int*   ei3      = (const int*)d_in[3];
    const int*   cluster1 = (const int*)d_in[4];
    const int*   cluster2 = (const int*)d_in[5];
    const float *cW1 = (const float*)d_in[6],  *cU1 = (const float*)d_in[7],
                *cc1 = (const float*)d_in[8],  *cb1 = (const float*)d_in[9];
    const float *cW2 = (const float*)d_in[10], *cU2 = (const float*)d_in[11],
                *cc2 = (const float*)d_in[12], *cb2 = (const float*)d_in[13];
    const float *cW3 = (const float*)d_in[14], *cU3 = (const float*)d_in[15],
                *cc3 = (const float*)d_in[16], *cb3 = (const float*)d_in[17];
    const float *cW4 = (const float*)d_in[18], *cU4 = (const float*)d_in[19],
                *cc4 = (const float*)d_in[20], *cb4 = (const float*)d_in[21];
    const float *cW5 = (const float*)d_in[22], *cU5 = (const float*)d_in[23],
                *cc5 = (const float*)d_in[24], *cb5 = (const float*)d_in[25];
    const float *l1W = (const float*)d_in[26], *l1b = (const float*)d_in[27];
    const float *l2W = (const float*)d_in[28], *l2b = (const float*)d_in[29];
    const float *oW  = (const float*)d_in[30], *ob  = (const float*)d_in[31];
    float* out = (float*)d_out;

    // ---- workspace carve-up (bytes, 256-aligned) ----
    char*  base = (char*)d_ws;
    size_t off  = 0;
    auto alloc = [&](size_t bytes) -> char* {
        char* p = base + off;
        off += (bytes + 255) & ~(size_t)255;
        return p;
    };
    float* m_buf  = (float*)alloc((size_t)2560000 * 4);  // max n*H*fo
    float* u_buf  = (float*)alloc((size_t)160000 * 4);   // max n*H
    float* agg    = (float*)alloc((size_t)640000 * 4);   // max n*fo
    float* cnt    = (float*)alloc((size_t)40000 * 4);
    float* x1     = (float*)alloc((size_t)N1 * 16 * 4);
    float* p1     = (float*)alloc((size_t)N2C * 16 * 4); // pool1 output (own slot!)
    float* x2     = (float*)alloc((size_t)N2C * 32 * 4);
    float* p2     = (float*)alloc((size_t)N3C * 32 * 4); // pool2 output (own slot!)
    float* x3a    = (float*)alloc((size_t)N3C * 64 * 4);
    float* x3b    = (float*)alloc((size_t)N3C * 32 * 4);
    float* x3cat  = (float*)alloc((size_t)N2C * 64 * 4);
    float* x5     = (float*)alloc((size_t)N2C * 16 * 4);
    float* h      = (float*)alloc((size_t)N1 * 32 * 4);
    unsigned short* a1 = (unsigned short*)alloc((size_t)N1 * 256 * 2);
    unsigned short* Bp = (unsigned short*)alloc((size_t)NCOLT * 8 * 32 * 16 * 2);
    float* l2bp   = (float*)alloc((size_t)NCOLP * 4);
    float* oWp    = (float*)alloc((size_t)NCOLP * 4);
    (void)ws_size; (void)in_sizes; (void)n_in; (void)out_size;

    // ---- graph stack ----
    run_conv(x,     N1,  5,  16, ei1, E1,  cW1, cU1, cc1, cb1, x1,   m_buf, u_buf, agg, cnt, stream);
    hipMemsetAsync(p1, 0, (size_t)N2C * 16 * sizeof(float), stream);
    seg_max_kernel<<<CDIV(N1 * 16, 256), 256, 0, stream>>>(x1, cluster1, p1, N1, 16);
    run_conv(p1,    N2C, 16, 32, ei2, E2C, cW2, cU2, cc2, cb2, x2,   m_buf, u_buf, agg, cnt, stream);
    hipMemsetAsync(p2, 0, (size_t)N3C * 32 * sizeof(float), stream);
    seg_max_kernel<<<CDIV(N2C * 32, 256), 256, 0, stream>>>(x2, cluster2, p2, N2C, 32);
    run_conv(p2,    N3C, 32, 64, ei3, E3C, cW3, cU3, cc3, cb3, x3a,  m_buf, u_buf, agg, cnt, stream);
    run_conv(x3a,   N3C, 64, 32, ei3, E3C, cW4, cU4, cc4, cb4, x3b,  m_buf, u_buf, agg, cnt, stream);
    concat_up_kernel<<<CDIV(N2C * 64, 256), 256, 0, stream>>>(x2, x3b, cluster2, x3cat, N2C, 32, 32);
    run_conv(x3cat, N2C, 64, 16, ei2, E2C, cW5, cU5, cc5, cb5, x5,   m_buf, u_buf, agg, cnt, stream);
    concat_up_kernel<<<CDIV(N1 * 32, 256), 256, 0, stream>>>(x1, x5, cluster1, h, N1, 16, 16);

    // ---- MLP head (WMMA) ----
    mlp1_kernel<<<CDIV(N1 * 256, 256), 256, 0, stream>>>(h, l1W, l1b, a1);
    pack_b_kernel<<<CDIV(NCOLT * 8 * 32 * 16, 256), 256, 0, stream>>>(l2W, Bp);
    pack_vec_kernel<<<CDIV(NCOLP, 256), 256, 0, stream>>>(l2b, oW, l2bp, oWp);
    fused_mlp2_kernel<<<CDIV(NROWT, WAVES_PER_BLOCK), 256, 0, stream>>>(a1, Bp, l2bp, oWp, ob, out);
}
